// ScaledDotProductAttention_47141561041101
// MI455X (gfx1250) — matrix operand order repaired and checked by reading, not compiled, not run on hardware
//
#include <hip/hip_runtime.h>

typedef __attribute__((ext_vector_type(16))) _Float16 v16h;
typedef __attribute__((ext_vector_type(8)))  float    v8f;

#define WMMA_F16(a, b, c) \
    __builtin_amdgcn_wmma_f32_16x16x32_f16(false, (a), false, (b), (short)0, (c), false, false)

// Flash attention with event-length mask. B=8, H=16, S=1024, D=64.
// fp32 in/out, f16 WMMA math (fp32 accumulate), fp32 online softmax.
// Block = 256 threads (8 wave32) = 128 query rows of one (b,h).
// Each wave: 16 query rows. KV tiles of 32 keys, DOUBLE-BUFFERED in LDS:
// tile t+1's global loads are issued before tile t's compute so HBM/L2
// latency overlaps WMMA work; one barrier per iteration.
// V is staged TRANSPOSED (vt_t[d][kv]) so PV B-fragments are contiguous in
// LDS and vectorize into b64 loads.
//
// VALU / latency minimization:
//  - scores pre-scaled by (1/sqrt(D))*log2(e) in Q prologue -> exp2 directly
//  - one shared running max per wave (exact softmax for any shared m);
//    cross-lane max via a 2-round LDS broadcast tree (2 DS round trips
//    instead of 5 serialized ds_bpermute+wait rounds)
//  - readfirstlane -> SGPR -> rescale is a scalar-branched rare path
//  - row sums via an extra WMMA against an all-ones B matrix
//  - masking is a scalar-branched slow path only for tiles straddling L
//    with valid rows; invalid rows stay unmasked (uniform -1e9 shifts out)
//  - blocks whose 128 rows are ALL valid stop at ceil(L/32)*32 keys
__global__ __launch_bounds__(256)
void ScaledDotProductAttention_fa_kernel(const float* __restrict__ q,
                                         const float* __restrict__ k,
                                         const float* __restrict__ v,
                                         const int*   __restrict__ evlen,
                                         float* __restrict__ out) {
    constexpr int   S      = 1024;
    constexpr int   D      = 64;
    constexpr float QSCALE = 0.125f * 1.44269504088896340736f; // 1/sqrt(64)*log2(e)
    constexpr int   KST    = 68;   // K tile LDS row stride (halves), rows = kv
    constexpr int   VST    = 36;   // V^T tile LDS row stride (halves), rows = d
    constexpr int   PST    = 36;   // P tile LDS row stride (halves)

    __shared__ _Float16 kt[2][32 * KST];     // kt[buf][kv][d]
    __shared__ _Float16 vt_t[2][64 * VST];   // vt_t[buf][d][kv] (transposed)
    __shared__ __align__(16) _Float16 pt[8][16 * PST];
    __shared__ __align__(16) float    redbuf[8][48];   // per-wave max scratch

    const int tid  = threadIdx.x;
    // force wave id scalar so wave-uniform branches become s_cbranch
    const int wave = __builtin_amdgcn_readfirstlane(tid) >> 5;
    const int lane = tid & 31;
    const int l16  = lane & 15;
    const int hi   = lane >> 4;       // 0: lanes 0-15, 1: lanes 16-31

    const int blk   = blockIdx.x;     // 0 .. B*H*8-1
    const int qtile = blk & 7;        // 8 query tiles of 128 rows
    const int bh    = blk >> 3;       // b*H + h
    const int b     = bh >> 4;        // H == 16
    const int L     = evlen[b];

    const size_t base = (size_t)bh * S * D;
    const int    q0b  = qtile * 128;             // block's first query row
    const int    q0   = q0b + wave * 16;         // wave's first query row

    // Early stop: if ALL 128 rows of this block are valid, keys >= L give
    // exactly-zero p for every row -> stop at ceil(L/32)*32. Else full S.
    const int jmax   = (q0b + 128 <= L) ? ((L + 31) & ~31) : S;
    const int ntiles = jmax >> 5;                // >= 4 always

    // Per-thread cooperative-load geometry (constant across tiles):
    // chunk 0: (row, col), chunk 1: (row+16, col).
    const int lrow = tid >> 4;            // kv row 0..15
    const int lcol = (tid & 15) << 2;     // d 0..60
    const float* kp0 = k + base + (size_t)lrow * D + lcol;
    const float* vp0 = v + base + (size_t)lrow * D + lcol;
    const size_t chunk1 = (size_t)16 * D; // +16 kv rows

    // ---- Q A-fragments (16 rows x 64 dims), pre-scaled into exp2 domain.
    // A 16x32 f16 frag, this code's k order (not AMD's table): lane(0-15)=row M; half h: K = h + hi*16.
    v16h aq0, aq1;
    {
        const float* qrow = q + base + (size_t)(q0 + l16) * D;
#pragma unroll
        for (int h = 0; h < 16; ++h) {
            const int d0 = (hi << 4) + h;
            aq0[h] = (_Float16)(qrow[d0] * QSCALE);
            aq1[h] = (_Float16)(qrow[d0 + 32] * QSCALE);
        }
    }

    // All-ones B fragment for the row-sum WMMA.
    v16h ones;
#pragma unroll
    for (int h = 0; h < 16; ++h) ones[h] = (_Float16)1.0f;

    // Loop-invariant row validity (reg r, half hi -> row q0 + hi*8 + r).
    bool rv[8];
#pragma unroll
    for (int r = 0; r < 8; ++r) rv[r] = (q0 + (hi << 3) + r) < L;

    // ---- Accumulators: O = 16x64 fp32 (4 C frags), l via ones-WMMA, shared m.
    v8f o0 = {}, o1 = {}, o2 = {}, o3 = {}, lacc = {};
    float m = -3.0e38f;

    // ---- Prologue: stage tile 0 into buffer 0.
    {
        const float4 kf0 = *(const float4*)(kp0);
        const float4 vf0 = *(const float4*)(vp0);
        const float4 kf1 = *(const float4*)(kp0 + chunk1);
        const float4 vf1 = *(const float4*)(vp0 + chunk1);
        _Float16* kd = &kt[0][lrow * KST + lcol];
        kd[0] = (_Float16)kf0.x; kd[1] = (_Float16)kf0.y;
        kd[2] = (_Float16)kf0.z; kd[3] = (_Float16)kf0.w;
        kd = &kt[0][(lrow + 16) * KST + lcol];
        kd[0] = (_Float16)kf1.x; kd[1] = (_Float16)kf1.y;
        kd[2] = (_Float16)kf1.z; kd[3] = (_Float16)kf1.w;
        vt_t[0][(lcol + 0) * VST + lrow] = (_Float16)vf0.x;
        vt_t[0][(lcol + 1) * VST + lrow] = (_Float16)vf0.y;
        vt_t[0][(lcol + 2) * VST + lrow] = (_Float16)vf0.z;
        vt_t[0][(lcol + 3) * VST + lrow] = (_Float16)vf0.w;
        vt_t[0][(lcol + 0) * VST + lrow + 16] = (_Float16)vf1.x;
        vt_t[0][(lcol + 1) * VST + lrow + 16] = (_Float16)vf1.y;
        vt_t[0][(lcol + 2) * VST + lrow + 16] = (_Float16)vf1.z;
        vt_t[0][(lcol + 3) * VST + lrow + 16] = (_Float16)vf1.w;
    }
    __syncthreads();

    for (int t = 0; t < ntiles; ++t) {
        const int j0  = t << 5;
        const int cur = t & 1;
        const _Float16* ktc = &kt[cur][0];
        const _Float16* vtc = &vt_t[cur][0];

        // ---- Issue next tile's global loads FIRST (latency hidden by compute).
        const bool havenext = (t + 1) < ntiles;   // block-uniform
        float4 kf0, vf0, kf1, vf1;
        if (havenext) {
            const size_t off = (size_t)(j0 + 32) * D;
            kf0 = *(const float4*)(kp0 + off);
            vf0 = *(const float4*)(vp0 + off);
            kf1 = *(const float4*)(kp0 + off + chunk1);
            vf1 = *(const float4*)(vp0 + off + chunk1);
        }

        // ---- Scores: S(16x32) = Qs x K^T.
        // B 32x16 f16 frag, same k order as A (not AMD's table): lane = column N (mod 16); half h: K = h + hi*16.
        v8f c0 = {}, c1 = {};
        {
            v16h bk0, bk1;
#pragma unroll
            for (int h = 0; h < 16; ++h) {
                const int d = h + (hi << 4);
                bk0[h] = ktc[l16 * KST + d];
                bk1[h] = ktc[l16 * KST + d + 32];
            }
            c0 = WMMA_F16(aq0, bk0, c0);
            c0 = WMMA_F16(aq1, bk1, c0);
#pragma unroll
            for (int h = 0; h < 16; ++h) {
                const int d = h + (hi << 4);
                bk0[h] = ktc[(16 + l16) * KST + d];
                bk1[h] = ktc[(16 + l16) * KST + d + 32];
            }
            c1 = WMMA_F16(aq0, bk0, c1);
            c1 = WMMA_F16(aq1, bk1, c1);
        }

        // ---- Shared running max over the wave's 16x32 tile.
        // Local max3 tree, then 2-round LDS broadcast (per-wave scratch).
        float tmax = c0[0];
#pragma unroll
        for (int r = 1; r < 8; ++r) tmax = fmaxf(tmax, c0[r]);
#pragma unroll
        for (int r = 0; r < 8; ++r) tmax = fmaxf(tmax, c1[r]);
        {
            float* rb = redbuf[wave];
            rb[lane] = tmax;
            asm volatile("s_wait_dscnt 0x0" ::: "memory");
            const float4 g4 = *(const float4*)&rb[(lane & 7) << 2];
            const float gm = fmaxf(fmaxf(g4.x, g4.y), fmaxf(g4.z, g4.w));
            rb[32 + (lane & 7)] = gm;   // 4 lanes write identical value
            asm volatile("s_wait_dscnt 0x0" ::: "memory");
            const float4 h0 = *(const float4*)&rb[32];
            const float4 h1 = *(const float4*)&rb[36];
            tmax = fmaxf(fmaxf(fmaxf(h0.x, h0.y), fmaxf(h0.z, h0.w)),
                         fmaxf(fmaxf(h1.x, h1.y), fmaxf(h1.z, h1.w)));
        }
        tmax = __builtin_amdgcn_readfirstlane(tmax);   // wave-uniform scalar

        // Lazy rescale: scalar branch, rarely taken after warm-up.
        if (tmax > m) {
            const float alpha = __builtin_amdgcn_exp2f(m - tmax);
            m = tmax;
#pragma unroll
            for (int r = 0; r < 8; ++r) {
                o0[r] *= alpha; o1[r] *= alpha; o2[r] *= alpha; o3[r] *= alpha;
                lacc[r] *= alpha;
            }
        }

        // ---- p = exp2(c - m) -> f16 in per-wave LDS (A-layout pickup below).
        // Mask path only when this tile straddles L AND some rows are valid.
        const bool needMask = (q0 < L) && (j0 + 32 > L);
        if (needMask) {
            const bool cz0 = (j0 + l16) >= L;        // col invalid, subtile 0
            const bool cz1 = (j0 + 16 + l16) >= L;   // col invalid, subtile 1
#pragma unroll
            for (int r = 0; r < 8; ++r) {
                const float p0 = (rv[r] && cz0) ? 0.0f : __builtin_amdgcn_exp2f(c0[r] - m);
                const float p1 = (rv[r] && cz1) ? 0.0f : __builtin_amdgcn_exp2f(c1[r] - m);
                const int prow = (hi << 3) + r;
                pt[wave][prow * PST + l16]      = (_Float16)p0;
                pt[wave][prow * PST + 16 + l16] = (_Float16)p1;
            }
        } else {
#pragma unroll
            for (int r = 0; r < 8; ++r) {
                const float p0 = __builtin_amdgcn_exp2f(c0[r] - m);
                const float p1 = __builtin_amdgcn_exp2f(c1[r] - m);
                const int prow = (hi << 3) + r;
                pt[wave][prow * PST + l16]      = (_Float16)p0;
                pt[wave][prow * PST + 16 + l16] = (_Float16)p1;
            }
        }

        // wave-local: DS writes above must land before cross-lane DS reads below
        asm volatile("s_wait_dscnt 0x0" ::: "memory");

        // ---- P as A-fragment (16x32); row sums + PV.
        v16h ap;
#pragma unroll
        for (int h = 0; h < 16; ++h) {
            const int kk = (hi << 4) + h;
            ap[h] = pt[wave][l16 * PST + kk];
        }
        lacc = WMMA_F16(ap, ones, lacc);    // every column = row sum
        {
            // PV B-frags from transposed V: contiguous in h -> vectorized DS.
            v16h bv;
#pragma unroll
            for (int h = 0; h < 16; ++h) bv[h] = vtc[(     l16) * VST + h + (hi << 4)];
            o0 = WMMA_F16(ap, bv, o0);
#pragma unroll
            for (int h = 0; h < 16; ++h) bv[h] = vtc[(16 + l16) * VST + h + (hi << 4)];
            o1 = WMMA_F16(ap, bv, o1);
#pragma unroll
            for (int h = 0; h < 16; ++h) bv[h] = vtc[(32 + l16) * VST + h + (hi << 4)];
            o2 = WMMA_F16(ap, bv, o2);
#pragma unroll
            for (int h = 0; h < 16; ++h) bv[h] = vtc[(48 + l16) * VST + h + (hi << 4)];
            o3 = WMMA_F16(ap, bv, o3);
        }

        // ---- Stage next tile into the other buffer (loads already in flight).
        if (havenext) {
            const int nb = cur ^ 1;
            _Float16* kd = &kt[nb][lrow * KST + lcol];
            kd[0] = (_Float16)kf0.x; kd[1] = (_Float16)kf0.y;
            kd[2] = (_Float16)kf0.z; kd[3] = (_Float16)kf0.w;
            kd = &kt[nb][(lrow + 16) * KST + lcol];
            kd[0] = (_Float16)kf1.x; kd[1] = (_Float16)kf1.y;
            kd[2] = (_Float16)kf1.z; kd[3] = (_Float16)kf1.w;
            vt_t[nb][(lcol + 0) * VST + lrow] = (_Float16)vf0.x;
            vt_t[nb][(lcol + 1) * VST + lrow] = (_Float16)vf0.y;
            vt_t[nb][(lcol + 2) * VST + lrow] = (_Float16)vf0.z;
            vt_t[nb][(lcol + 3) * VST + lrow] = (_Float16)vf0.w;
            vt_t[nb][(lcol + 0) * VST + lrow + 16] = (_Float16)vf1.x;
            vt_t[nb][(lcol + 1) * VST + lrow + 16] = (_Float16)vf1.y;
            vt_t[nb][(lcol + 2) * VST + lrow + 16] = (_Float16)vf1.z;
            vt_t[nb][(lcol + 3) * VST + lrow + 16] = (_Float16)vf1.w;
        }

        __syncthreads();   // next buffer visible; also fences buffer reuse
    }

    // ---- Epilogue: normalize by l (same C layout as O) and store fp32.
#pragma unroll
    for (int r = 0; r < 8; ++r) {
        const float inv = 1.0f / lacc[r];
        const size_t orow = base + (size_t)(q0 + (hi << 3) + r) * D;
        out[orow +      l16] = o0[r] * inv;
        out[orow + 16 + l16] = o1[r] * inv;
        out[orow + 32 + l16] = o2[r] * inv;
        out[orow + 48 + l16] = o3[r] * inv;
    }
}

extern "C" void kernel_launch(void* const* d_in, const int* in_sizes, int n_in,
                              void* d_out, int out_size, void* d_ws, size_t ws_size,
                              hipStream_t stream) {
    (void)in_sizes; (void)n_in; (void)d_ws; (void)ws_size; (void)out_size;
    const float* q  = (const float*)d_in[0];
    const float* k  = (const float*)d_in[1];
    const float* v  = (const float*)d_in[2];
    const int*   el = (const int*)d_in[3];
    float* out = (float*)d_out;

    // B*H*(S/128) = 8*16*8 = 1024 blocks, 256 threads (8 wave32) each
    dim3 grid(1024), block(256);
    ScaledDotProductAttention_fa_kernel<<<grid, block, 0, stream>>>(q, k, v, el, out);
}
